// CommandScorerWithKG_30872224923914
// MI455X (gfx1250) — matrix operand order repaired and checked by reading, not compiled, not run on hardware
//
#include <hip/hip_runtime.h>
#include <hip/hip_bf16.h>

typedef __bf16 bf16_t;
typedef __attribute__((ext_vector_type(16))) __bf16 v16bf;
typedef __attribute__((ext_vector_type(8)))  __bf16 v8bf;
typedef __attribute__((ext_vector_type(8)))  float  v8f;

// Pointer types for the async global->LDS builtin:
// param 0 is AS(1) ptr to 16-byte int vector, param 1 is AS(3) ditto.
typedef __attribute__((__vector_size__(16))) int i32x4_t;
typedef __attribute__((address_space(1))) i32x4_t g_i32x4;
typedef __attribute__((address_space(3))) i32x4_t l_i32x4;

#define B_   32
#define L_   2048
#define DW_  300
#define DH_  100
#define KP_  416          // projection K padded 400 -> 416 (13 * 32)
#define H_   128
#define G3_  384          // 3*H

#if defined(__has_builtin)
#if __has_builtin(__builtin_amdgcn_global_load_async_to_lds_b128)
#define HAS_ASYNC_LDS 1
#endif
#if __has_builtin(__builtin_amdgcn_s_wait_asynccnt)
#define HAS_WAIT_ASYNC 1
#endif
#endif

// ---------------------------------------------------------------------------
// WMMA fragment loaders per CDNA5 ISA layouts (wave32).
// A (16-bit, 16x32 MxK): lanes 0-15 / 16-31 both hold rows M=0..15;
//   element i<8:  K = 8*half + i ; element i>=8: K = 16 + 8*half + (i-8)
// B (16-bit, 32x16 KxN): same K order as A, N = lane & 15
//   (weights stored [N][K] so each fragment is two contiguous b128 loads)
// ---------------------------------------------------------------------------
__device__ __forceinline__ v16bf frag_a(const bf16_t* base, int row, int k0,
                                        int half, int ldk) {
  const bf16_t* p = base + row * ldk + k0 + half * 8;
  v8bf lo = *(const v8bf*)p;          // K = k0 + 8*half + [0..7]
  v8bf hi = *(const v8bf*)(p + 16);   // K = k0 + 16 + 8*half + [0..7]
  v16bf r;
#pragma unroll
  for (int i = 0; i < 8; ++i) { r[i] = lo[i]; r[i + 8] = hi[i]; }
  return r;
}

__device__ __forceinline__ v16bf frag_b(const bf16_t* baseNK, int n, int k0,
                                        int half, int ldk) {
  const bf16_t* p = baseNK + n * ldk + k0 + half * 8;   // same K order as frag_a
  v8bf lo = *(const v8bf*)p;          // K = k0 + 8*half + [0..7]
  v8bf hi = *(const v8bf*)(p + 16);   // K = k0 + 16 + 8*half + [0..7]
  v16bf r;
#pragma unroll
  for (int i = 0; i < 8; ++i) { r[i] = lo[i]; r[i + 8] = hi[i]; }
  return r;
}

// Async-copy one 16-byte chunk from global to LDS (ASYNCcnt path).
__device__ __forceinline__ void async_copy_b128(const bf16_t* gsrc, bf16_t* ldst,
                                                int chunk) {
#ifdef HAS_ASYNC_LDS
  __builtin_amdgcn_global_load_async_to_lds_b128(
      (g_i32x4*)(gsrc + chunk * 8), (l_i32x4*)(ldst + chunk * 8), 0, 0);
#else
  *(ulong2*)(ldst + chunk * 8) = *(const ulong2*)(gsrc + chunk * 8);
#endif
}

__device__ __forceinline__ void wait_async() {
#ifdef HAS_ASYNC_LDS
#ifdef HAS_WAIT_ASYNC
  __builtin_amdgcn_s_wait_asynccnt(0);
#else
  asm volatile("s_wait_asynccnt 0x0" ::: "memory");
#endif
#endif
}

// ---------------------------------------------------------------------------
// Phase 1: gather embeddings, project through W_prj, emit x[L][B][H] as bf16.
// One block handles 8 timesteps; W_prj cached in LDS transposed [N=128][K=416].
// ---------------------------------------------------------------------------
__global__ __launch_bounds__(256, 1)
void proj_kernel(const int* __restrict__ obs, const float* __restrict__ mask,
                 const int* __restrict__ nb2hyp, const float* __restrict__ wtab,
                 const float* __restrict__ htab, const float* __restrict__ Wprj,
                 bf16_t* __restrict__ xout) {
  extern __shared__ char smem[];
  bf16_t* WpL  = (bf16_t*)smem;               // [128][416] bf16 = 106496 B
  bf16_t* AtL  = (bf16_t*)(smem + 106496);    // [32][416]  bf16 =  26624 B
  int*    tokL = (int*)  (smem + 133120);     // [32]
  int*    hypL = (int*)  (smem + 133248);     // [32]
  float*  mskL = (float*)(smem + 133376);     // [32]

  const int tid  = threadIdx.x;
  const int lane = tid & 31, wave = tid >> 5, half = lane >> 4;

  // W_prj is [400][128] row-major; store transposed [n][k], zero-pad K.
  for (int i = tid; i < H_ * KP_; i += 256) {
    int n = i / KP_, k = i % KP_;
    float v = (k < DW_ + DH_) ? Wprj[k * H_ + n] : 0.f;
    WpL[i] = (bf16_t)v;
  }
  __syncthreads();

  const int l0 = blockIdx.x * 8;
  for (int li = 0; li < 8; ++li) {
    const int l = l0 + li;
    if (tid < B_) {
      int tok   = obs[tid * L_ + l];
      tokL[tid] = tok;
      hypL[tid] = nb2hyp[tok];
      mskL[tid] = mask[tid * L_ + l];
    }
    __syncthreads();

    // Gather one 32 x 416 activation tile (concat word | mask*hyp | pad).
    for (int i = tid; i < B_ * KP_; i += 256) {
      int b = i / KP_, k = i % KP_;
      float v;
      if (k < DW_)            v = wtab[(long)tokL[b] * DW_ + k];
      else if (k < DW_ + DH_) v = htab[(long)hypL[b] * DH_ + (k - DW_)] * mskL[b];
      else                    v = 0.f;
      AtL[i] = (bf16_t)v;
    }
    __syncthreads();

    // 16 output tiles (2 Mtiles x 8 Ntiles); each wave: 1 Mtile x 2 Ntiles,
    // sharing the A fragment and running two independent WMMA chains.
    {
      const int mt = wave >> 2;          // waves 0-3 -> mt 0, 4-7 -> mt 1
      const int n0 = (wave & 3) * 2;     // two consecutive N tiles
      v8f acc0 = {}, acc1 = {};
#pragma unroll
      for (int kt = 0; kt < 13; ++kt) {
        v16bf a  = frag_a(AtL, mt * 16 + (lane & 15), kt * 32, half, KP_);
        v16bf b0 = frag_b(WpL, (n0 + 0) * 16 + (lane & 15), kt * 32, half, KP_);
        acc0 = __builtin_amdgcn_wmma_f32_16x16x32_bf16(false, a, false, b0,
                                                       (short)0, acc0, false, false);
        v16bf b1 = frag_b(WpL, (n0 + 1) * 16 + (lane & 15), kt * 32, half, KP_);
        acc1 = __builtin_amdgcn_wmma_f32_16x16x32_bf16(false, a, false, b1,
                                                       (short)0, acc1, false, false);
      }
#pragma unroll
      for (int r = 0; r < 8; ++r) {
        int m = mt * 16 + r + half * 8;   // C/D layout: VGPR r -> M=r / r+8
        int nA = (n0 + 0) * 16 + (lane & 15);
        int nB = (n0 + 1) * 16 + (lane & 15);
        xout[((long)l * B_ + m) * H_ + nA] = (bf16_t)acc0[r];
        xout[((long)l * B_ + m) * H_ + nB] = (bf16_t)acc1[r];
      }
    }
    __syncthreads();
  }
}

// ---------------------------------------------------------------------------
// Phase 2: sequential GRU, one persistent workgroup per direction (grid = 2).
// 24 waves (768 threads). Weights live in LDS as bf16 (192 KB); x_t is
// double-buffered in LDS via GLOBAL_LOAD_ASYNC_TO_LDS_B128 (ASYNCcnt).
// Waves 0-11: gi = x_t @ Wih^T ; waves 12-23: gh = h @ Whh^T.
// ---------------------------------------------------------------------------
__global__ __launch_bounds__(768, 1)
void gru_kernel(const bf16_t* __restrict__ x,
                const float* __restrict__ Wih_f, const float* __restrict__ Whh_f,
                const float* __restrict__ bih_f, const float* __restrict__ bhh_f,
                const float* __restrict__ Wih_b, const float* __restrict__ Whh_b,
                const float* __restrict__ bih_b, const float* __restrict__ bhh_b,
                float* __restrict__ out) {
  extern __shared__ char smem[];
  bf16_t* WihL = (bf16_t*)smem;               // [384][128] bf16 =  98304 B
  bf16_t* WhhL = (bf16_t*)(smem + 98304);     // [384][128] bf16 =  98304 B
  float*  giL  = (float*) (smem + 196608);    // [32][384]  f32  =  49152 B
  float*  ghL  = (float*) (smem + 245760);    // [32][384]  f32  =  49152 B
  bf16_t* hbf  = (bf16_t*)(smem + 294912);    // [32][128]  bf16 =   8192 B
  bf16_t* xstg = (bf16_t*)(smem + 303104);    // 2 x [32][128] bf16 = 16384 B
                                              // total 319488 B < 320 KB LDS

  const int dir = blockIdx.x;
  const float* Wih = dir ? Wih_b : Wih_f;
  const float* Whh = dir ? Whh_b : Whh_f;
  const float* bih = dir ? bih_b : bih_f;
  const float* bhh = dir ? bhh_b : bhh_f;

  const int tid  = threadIdx.x;
  const int lane = tid & 31, wave = tid >> 5, half = lane >> 4;

  // PyTorch Wih/Whh are [3H][H] row-major == exactly our [N][K] B layout.
  for (int i = tid; i < G3_ * H_; i += 768) {
    WihL[i] = (bf16_t)Wih[i];
    WhhL[i] = (bf16_t)Whh[i];
  }
  for (int i = tid; i < B_ * H_; i += 768) hbf[i] = (bf16_t)0.f;

  // Prologue: stage x for t=0 into buffer 0 (512 x 16-byte chunks).
  {
    const bf16_t* x0 = x + (long)(dir ? L_ - 1 : 0) * B_ * H_;
    for (int c = tid; c < 512; c += 768) async_copy_b128(x0, xstg, c);
  }
  wait_async();
  __syncthreads();

  const bool isG = wave < 12;
  const int  gw  = isG ? wave : wave - 12;    // 0..11 within GEMM group
  const int  mt  = gw / 6;                    // 2 Mtiles
  const int  n0  = (gw % 6) * 4;              // 4 consecutive Ntiles
  const bf16_t* Wm  = isG ? WihL : WhhL;
  float*        Dst = isG ? giL  : ghL;

  for (int t = 0; t < L_; ++t) {
    const int cur = t & 1, nxt = cur ^ 1;
    const bf16_t* xcur = xstg + cur * (B_ * H_);

    // Kick async staging of the next timestep while this step's WMMAs run.
    if (t + 1 < L_) {
      const int ln = dir ? (L_ - 2 - t) : (t + 1);
      const bf16_t* xn = x + (long)ln * B_ * H_;
      bf16_t* dstb = xstg + nxt * (B_ * H_);
      for (int c = tid; c < 512; c += 768) async_copy_b128(xn, dstb, c);
    }

    // GEMM: 48 tiles each (2 Mtiles x 24 Ntiles), K=128 -> 4 WMMA per tile.
    // One A fragment per kt feeds 4 independent accumulator chains.
    {
      const bf16_t* A = isG ? xcur : hbf;
      v8f a0 = {}, a1 = {}, a2 = {}, a3 = {};
#pragma unroll
      for (int kt = 0; kt < 4; ++kt) {
        v16bf af = frag_a(A, mt * 16 + (lane & 15), kt * 32, half, H_);
        v16bf b0 = frag_b(Wm, (n0 + 0) * 16 + (lane & 15), kt * 32, half, H_);
        a0 = __builtin_amdgcn_wmma_f32_16x16x32_bf16(false, af, false, b0,
                                                     (short)0, a0, false, false);
        v16bf b1 = frag_b(Wm, (n0 + 1) * 16 + (lane & 15), kt * 32, half, H_);
        a1 = __builtin_amdgcn_wmma_f32_16x16x32_bf16(false, af, false, b1,
                                                     (short)0, a1, false, false);
        v16bf b2 = frag_b(Wm, (n0 + 2) * 16 + (lane & 15), kt * 32, half, H_);
        a2 = __builtin_amdgcn_wmma_f32_16x16x32_bf16(false, af, false, b2,
                                                     (short)0, a2, false, false);
        v16bf b3 = frag_b(Wm, (n0 + 3) * 16 + (lane & 15), kt * 32, half, H_);
        a3 = __builtin_amdgcn_wmma_f32_16x16x32_bf16(false, af, false, b3,
                                                     (short)0, a3, false, false);
      }
#pragma unroll
      for (int r = 0; r < 8; ++r) {
        const int m = (mt * 16 + r + half * 8) * G3_ + (lane & 15);
        Dst[m + (n0 + 0) * 16] = a0[r];
        Dst[m + (n0 + 1) * 16] = a1[r];
        Dst[m + (n0 + 2) * 16] = a2[r];
        Dst[m + (n0 + 3) * 16] = a3[r];
      }
    }
    __syncthreads();

    // Gate math: PyTorch order (r, z, n);  n = tanh(inn + r * hn).
    for (int i = tid; i < B_ * H_; i += 768) {
      const int b = i >> 7, j = i & 127;
      const float ir = giL[b * G3_ + j]       + bih[j];
      const float iz = giL[b * G3_ + 128 + j] + bih[128 + j];
      const float in = giL[b * G3_ + 256 + j] + bih[256 + j];
      const float hr = ghL[b * G3_ + j]       + bhh[j];
      const float hz = ghL[b * G3_ + 128 + j] + bhh[128 + j];
      const float hn = ghL[b * G3_ + 256 + j] + bhh[256 + j];
      const float rg = 1.f / (1.f + __expf(-(ir + hr)));
      const float zg = 1.f / (1.f + __expf(-(iz + hz)));
      const float ng = tanhf(in + rg * hn);
      const float hnew = (1.f - zg) * ng + zg * (float)hbf[i];
      hbf[i] = (bf16_t)hnew;
    }
    wait_async();        // next-step staging complete before the barrier
    __syncthreads();
  }

  // enc layout: out[b*257 + 0..127] = hf ; out[b*257 + 128..255] = hb
  for (int i = tid; i < B_ * H_; i += 768) {
    const int b = i >> 7, j = i & 127;
    out[b * 257 + dir * H_ + j] = (float)hbf[i];
  }
}

// ---------------------------------------------------------------------------
// Phase 3: critic head  value[b] = enc[b,:256] . Wc + bc
// ---------------------------------------------------------------------------
__global__ __launch_bounds__(256, 1)
void value_kernel(const float* __restrict__ Wc, const float* __restrict__ bc,
                  float* __restrict__ out) {
  __shared__ float red[256];
  const int tid = threadIdx.x;
  const int b = tid >> 3, s = tid & 7;
  float p = 0.f;
  for (int j = s * 32; j < s * 32 + 32; ++j) p += out[b * 257 + j] * Wc[j];
  red[tid] = p;
  __syncthreads();
  if (s == 0) {
    float v = bc[0];
    for (int k = 0; k < 8; ++k) v += red[b * 8 + k];
    out[b * 257 + 256] = v;
  }
}

extern "C" void kernel_launch(void* const* d_in, const int* in_sizes, int n_in,
                              void* d_out, int out_size, void* d_ws, size_t ws_size,
                              hipStream_t stream) {
  (void)in_sizes; (void)n_in; (void)out_size; (void)ws_size;
  const int*   obs    = (const int*)  d_in[0];
  const float* mask   = (const float*)d_in[1];
  const int*   nb2hyp = (const int*)  d_in[2];
  const float* wtab   = (const float*)d_in[3];
  const float* htab   = (const float*)d_in[4];
  const float* Wprj   = (const float*)d_in[5];
  const float* Wih_f  = (const float*)d_in[6];
  const float* Whh_f  = (const float*)d_in[7];
  const float* bih_f  = (const float*)d_in[8];
  const float* bhh_f  = (const float*)d_in[9];
  const float* Wih_b  = (const float*)d_in[10];
  const float* Whh_b  = (const float*)d_in[11];
  const float* bih_b  = (const float*)d_in[12];
  const float* bhh_b  = (const float*)d_in[13];
  const float* Wc     = (const float*)d_in[14];
  const float* bc     = (const float*)d_in[15];
  float*  out  = (float*)d_out;
  bf16_t* xbuf = (bf16_t*)d_ws;   // x[L][B][H] bf16 = 16 MB scratch

  // Allow >64KB dynamic LDS (host-side attribute set; deterministic).
  (void)hipFuncSetAttribute((const void*)proj_kernel,
                            hipFuncAttributeMaxDynamicSharedMemorySize, 133504);
  (void)hipFuncSetAttribute((const void*)gru_kernel,
                            hipFuncAttributeMaxDynamicSharedMemorySize, 319488);

  proj_kernel<<<L_ / 8, 256, 133504, stream>>>(obs, mask, nb2hyp, wtab, htab,
                                               Wprj, xbuf);
  gru_kernel<<<2, 768, 319488, stream>>>(xbuf, Wih_f, Whh_f, bih_f, bhh_f,
                                         Wih_b, Whh_b, bih_b, bhh_b, out);
  value_kernel<<<1, 256, 0, stream>>>(Wc, bc, out);
}
